// FusedGatedDeltaNet_72696616452197
// MI455X (gfx1250) — matrix operand order repaired and checked by reading, not compiled, not run on hardware
//
#include <hip/hip_runtime.h>
#include <hip/hip_bf16.h>

typedef __attribute__((ext_vector_type(16))) __bf16 v16bf;
typedef __attribute__((ext_vector_type(8)))  __bf16 v8bf;
typedef __attribute__((ext_vector_type(8)))  float  v8f;

#define S_LEN     1024
#define D_IN      2048
#define H_QK      8
#define H_V       16
#define DK        128
#define DV        128
#define FUSED_C   4096
#define D_OUT_VG  2048
#define EPSF      1e-6f

// ---------- helpers ----------
__device__ __forceinline__ __bf16 f2bf(float f) {
  unsigned u = __float_as_uint(f);
  unsigned r = (u + 0x7FFFu + ((u >> 16) & 1u)) >> 16;
  unsigned short h = (unsigned short)r;
  __bf16 b;
  __builtin_memcpy(&b, &h, 2);
  return b;
}

__device__ __forceinline__ float sigmoidf_(float x) { return 1.f / (1.f + __expf(-x)); }
__device__ __forceinline__ float siluf_(float x)    { return x * sigmoidf_(x); }

// ---------- fp32 -> bf16 conversion ----------
__global__ void k_f32_to_bf16(const float* __restrict__ s, __bf16* __restrict__ d, int n) {
  int i = blockIdx.x * blockDim.x + threadIdx.x;
  if (i < n) d[i] = f2bf(s[i]);
}

// ---------- bf16 WMMA GEMM: C[M,N] = A[M,K] * W[N,K]^T ----------
// A fragment (16x32 bf16): lane l holds row (l&15), K chunks [h8,h8+8) and [h8+16,h8+24), h8=(l>>4)*8
// B fragment (32x16 bf16): lane l holds col (l&15), K chunks [h8,h8+8) and [h8+16,h8+24), h8=(l>>4)*8
// C/D: VGPR r <-> (m = r + 8*(l>>4), n = l&15)
__device__ __forceinline__ v16bf load_a_frag(const __bf16* __restrict__ p) {
  v8bf lo = *(const v8bf*)(p);
  v8bf hi = *(const v8bf*)(p + 16);
  v16bf r;
#pragma unroll
  for (int i = 0; i < 8; ++i) { r[i] = lo[i]; r[i + 8] = hi[i]; }
  return r;
}

template <int WN>
__global__ void __launch_bounds__(256)
k_gemm_bf16_wmma(const __bf16* __restrict__ A, const __bf16* __restrict__ W,
                 float* __restrict__ C, int M, int N, int K, int tasks) {
  int wid  = threadIdx.x >> 5;
  int lane = threadIdx.x & 31;
  int task = blockIdx.x * (blockDim.x >> 5) + wid;
  if (task >= tasks) return;

  int ntN = N / (16 * WN);
  int tm  = task / ntN;
  int tn0 = (task % ntN) * WN;

  int mrow  = lane & 15;
  int half8 = (lane >> 4) * 8;
  int kh16  = (lane >> 4) * 16;

  const __bf16* aptr = A + (size_t)(tm * 16 + mrow) * K + half8;
  const __bf16* bptr[WN];
#pragma unroll
  for (int w = 0; w < WN; ++w)
    bptr[w] = W + (size_t)((tn0 + w) * 16 + mrow) * K + half8;

  v8f acc[WN];
#pragma unroll
  for (int w = 0; w < WN; ++w)
#pragma unroll
    for (int r = 0; r < 8; ++r) acc[w][r] = 0.f;

  for (int k0 = 0; k0 < K; k0 += 32) {
    __builtin_prefetch((const void*)(aptr + k0 + 256), 0, 1);  // global_prefetch_b8
    v16bf a = load_a_frag(aptr + k0);
#pragma unroll
    for (int w = 0; w < WN; ++w) {
      v16bf b = load_a_frag(bptr[w] + k0);
      acc[w] = __builtin_amdgcn_wmma_f32_16x16x32_bf16(
          false, a, false, b, (short)0, acc[w], false, false);
    }
  }

#pragma unroll
  for (int w = 0; w < WN; ++w) {
    int n = (tn0 + w) * 16 + (lane & 15);
#pragma unroll
    for (int r = 0; r < 8; ++r) {
      int m = tm * 16 + r + (lane >> 4) * 8;
      C[(size_t)m * N + n] = acc[w][r];
    }
  }
}

// ---------- causal depthwise conv (K=4) + SiLU ----------
__global__ void k_conv_silu(const float* __restrict__ qkv, const float* __restrict__ cw,
                            float* __restrict__ out) {
  int idx = blockIdx.x * blockDim.x + threadIdx.x;
  if (idx >= S_LEN * FUSED_C) return;
  int s = idx >> 12;
  int c = idx & (FUSED_C - 1);
  float acc = 0.f;
#pragma unroll
  for (int j = 0; j < 4; ++j) {
    int sp = s - 3 + j;
    if (sp >= 0) acc += qkv[(size_t)sp * FUSED_C + c] * cw[c * 4 + j];
  }
  out[idx] = siluf_(acc);
}

// ---------- l2-norm q,k in place + beta/alpha scalars ----------
__global__ void __launch_bounds__(512)
k_prep(float* __restrict__ fused, const float* __restrict__ betap,
       const float* __restrict__ alphap, const float* __restrict__ logA,
       const float* __restrict__ dtb, float* __restrict__ betas,
       float* __restrict__ alphas) {
  int s = blockIdx.x;
  int wid = threadIdx.x >> 5, lane = threadIdx.x & 31;
  // wid 0..7 -> q head wid; wid 8..15 -> k head (wid-8); contiguous at wid*128
  float* row = fused + (size_t)s * FUSED_C + wid * 128;
  float x0 = row[lane], x1 = row[lane + 32], x2 = row[lane + 64], x3 = row[lane + 96];
  float ss = x0 * x0 + x1 * x1 + x2 * x2 + x3 * x3;
#pragma unroll
  for (int off = 16; off >= 1; off >>= 1) ss += __shfl_xor(ss, off, 32);
  float inv = 1.f / sqrtf(ss + EPSF);
  row[lane] = x0 * inv; row[lane + 32] = x1 * inv;
  row[lane + 64] = x2 * inv; row[lane + 96] = x3 * inv;

  if (threadIdx.x < H_V) {
    int h = threadIdx.x;
    betas[h * S_LEN + s] = sigmoidf_(betap[s * H_V + h]);
    float tp = alphap[s * H_V + h] + dtb[h];
    float sp = (tp > 20.f) ? tp : log1pf(__expf(tp));
    alphas[h * S_LEN + s] = __expf(-__expf(logA[h]) * sp);
  }
}

// ---------- gated delta rule scan: 1 block per v-head, lane t owns state column S[:,t] ----------
__global__ void __launch_bounds__(128, 1)
k_scan(const float* __restrict__ fused, const float* __restrict__ betas,
       const float* __restrict__ alphas, float* __restrict__ ctx) {
  int h = blockIdx.x, hq = h >> 1, t = threadIdx.x;
  __shared__ float qs[DK], ks[DK], sc[2];
  float S[DK];
#pragma unroll
  for (int i = 0; i < DK; ++i) S[i] = 0.f;

  for (int s = 0; s < S_LEN; ++s) {
    const float* base = fused + (size_t)s * FUSED_C;
    qs[t] = base[hq * DK + t];
    ks[t] = base[1024 + hq * DK + t];
    if (t == 0) { sc[0] = alphas[h * S_LEN + s]; sc[1] = betas[h * S_LEN + s]; }
    __syncthreads();
    float a = sc[0], bt = sc[1];
    float vt = base[2048 + h * DV + t];
    float kv = 0.f;
#pragma unroll
    for (int k = 0; k < DK; ++k) { float sk = S[k] * a; S[k] = sk; kv += ks[k] * sk; }
    float u = (vt - kv) * bt;
    float o = 0.f;
#pragma unroll
    for (int k = 0; k < DK; ++k) { float sk = S[k] + ks[k] * u; S[k] = sk; o += qs[k] * sk; }
    ctx[((size_t)h * S_LEN + s) * DV + t] = o * 0.08838834764831845f;  // * dk^-0.5
    __syncthreads();
  }
}

// ---------- RMS-norm(ctx) * rms_w * silu(gate) -> bf16 G ----------
__global__ void __launch_bounds__(256)
k_finalize(const float* __restrict__ ctx, const float* __restrict__ gatep,
           const float* __restrict__ rmsw, __bf16* __restrict__ Gb) {
  int wid = threadIdx.x >> 5, lane = threadIdx.x & 31;
  int row = blockIdx.x * 8 + wid;         // row = h*1024 + s
  int h = row >> 10, s = row & 1023;
  const float* cr = ctx + (size_t)row * DV;
  float c0 = cr[lane], c1 = cr[lane + 32], c2 = cr[lane + 64], c3 = cr[lane + 96];
  float ss = c0 * c0 + c1 * c1 + c2 * c2 + c3 * c3;
#pragma unroll
  for (int off = 16; off >= 1; off >>= 1) ss += __shfl_xor(ss, off, 32);
  float inv = rsqrtf(ss * (1.f / DV) + EPSF);
  size_t ob = (size_t)s * D_OUT_VG + h * DV;
  float v[4] = {c0, c1, c2, c3};
#pragma unroll
  for (int j = 0; j < 4; ++j) {
    int d = lane + 32 * j;
    float g = gatep[ob + d];
    Gb[ob + d] = f2bf(siluf_(g) * (v[j] * inv * rmsw[d]));
  }
}

// ---------- launch ----------
extern "C" void kernel_launch(void* const* d_in, const int* in_sizes, int n_in,
                              void* d_out, int out_size, void* d_ws, size_t ws_size,
                              hipStream_t stream) {
  const float* x       = (const float*)d_in[0];
  const float* w_qkv   = (const float*)d_in[1];
  const float* w_gate  = (const float*)d_in[2];
  const float* w_beta  = (const float*)d_in[3];
  const float* w_alpha = (const float*)d_in[4];
  const float* log_A   = (const float*)d_in[5];
  const float* dt_bias = (const float*)d_in[6];
  const float* conv_w  = (const float*)d_in[7];
  const float* rms_w   = (const float*)d_in[8];
  const float* w_out   = (const float*)d_in[9];

  char* ws = (char*)d_ws;
  size_t off = 0;
  auto alloc = [&](size_t bytes) -> void* {
    void* p = ws + off;
    off += (bytes + 255) & ~(size_t)255;
    return p;
  };

  __bf16* xb      = (__bf16*)alloc((size_t)S_LEN * D_IN * 2);
  __bf16* wqkvb   = (__bf16*)alloc((size_t)FUSED_C * D_IN * 2);
  __bf16* wgateb  = (__bf16*)alloc((size_t)D_OUT_VG * D_IN * 2);
  __bf16* woutb   = (__bf16*)alloc((size_t)D_IN * D_OUT_VG * 2);
  __bf16* wbetab  = (__bf16*)alloc((size_t)H_V * D_IN * 2);
  __bf16* walphab = (__bf16*)alloc((size_t)H_V * D_IN * 2);
  float*  qkvf    = (float*)alloc((size_t)S_LEN * FUSED_C * 4);
  float*  gatef   = (float*)alloc((size_t)S_LEN * D_OUT_VG * 4);
  float*  betap   = (float*)alloc((size_t)S_LEN * H_V * 4);
  float*  alphap  = (float*)alloc((size_t)S_LEN * H_V * 4);
  float*  fusedc  = (float*)alloc((size_t)S_LEN * FUSED_C * 4);
  float*  betas   = (float*)alloc((size_t)H_V * S_LEN * 4);
  float*  alphas  = (float*)alloc((size_t)H_V * S_LEN * 4);
  float*  ctx     = (float*)alloc((size_t)H_V * S_LEN * DV * 4);
  __bf16* Gb      = (__bf16*)alloc((size_t)S_LEN * D_OUT_VG * 2);
  (void)ws_size; (void)in_sizes; (void)n_in; (void)out_size;

  auto cvt = [&](const float* s, __bf16* d, int n) {
    k_f32_to_bf16<<<(n + 255) / 256, 256, 0, stream>>>(s, d, n);
  };
  cvt(x,       xb,      S_LEN * D_IN);
  cvt(w_qkv,   wqkvb,   FUSED_C * D_IN);
  cvt(w_gate,  wgateb,  D_OUT_VG * D_IN);
  cvt(w_out,   woutb,   D_IN * D_OUT_VG);
  cvt(w_beta,  wbetab,  H_V * D_IN);
  cvt(w_alpha, walphab, H_V * D_IN);

  auto gemm4 = [&](const __bf16* A, const __bf16* W, float* C, int M, int N, int K) {
    int tasks = (M / 16) * (N / 64);
    k_gemm_bf16_wmma<4><<<(tasks + 7) / 8, 256, 0, stream>>>(A, W, C, M, N, K, tasks);
  };
  auto gemm1 = [&](const __bf16* A, const __bf16* W, float* C, int M, int N, int K) {
    int tasks = (M / 16) * (N / 16);
    k_gemm_bf16_wmma<1><<<(tasks + 7) / 8, 256, 0, stream>>>(A, W, C, M, N, K, tasks);
  };

  gemm4(xb, wqkvb,   qkvf,   S_LEN, FUSED_C,  D_IN);   // fused qkv proj
  gemm4(xb, wgateb,  gatef,  S_LEN, D_OUT_VG, D_IN);   // gate proj
  gemm1(xb, wbetab,  betap,  S_LEN, H_V,      D_IN);   // beta proj
  gemm1(xb, walphab, alphap, S_LEN, H_V,      D_IN);   // alpha proj

  k_conv_silu<<<(S_LEN * FUSED_C + 255) / 256, 256, 0, stream>>>(qkvf, conv_w, fusedc);

  k_prep<<<S_LEN, 512, 0, stream>>>(fusedc, betap, alphap, log_A, dt_bias, betas, alphas);

  k_scan<<<H_V, 128, 0, stream>>>(fusedc, betas, alphas, ctx);

  k_finalize<<<(H_V * S_LEN) / 8, 256, 0, stream>>>(ctx, gatef, rms_w, Gb);

  gemm4(Gb, woutb, (float*)d_out, S_LEN, D_IN, D_OUT_VG);  // output proj
}
